// PointNet2Model_56916906606965
// MI455X (gfx1250) — compile-verified, matrix operand order checked by reading, not run on hardware
//
#include <hip/hip_runtime.h>
#include <math.h>

typedef __attribute__((ext_vector_type(16))) _Float16 v16h;
typedef __attribute__((ext_vector_type(8)))  float    v8f;

#define TILE_M 256
#define TILE_N 64
#define TILE_K 32
#define LDSK   40   // halfs per row: 80B stride -> every fragment read is 16B aligned

__device__ __forceinline__ float silu_f(float x) { return x / (1.0f + expf(-x)); }

// ---------------------------------------------------------------------------
// WMMA GEMM: C[M,N] = A[M,Kp](f32, K zero-padded to Kp) @ Wt + bias (+SiLU).
// Wt is pre-transposed & padded f16: Wt[N][Kp] (n-major, k contiguous).
// Requirements (guaranteed by host orchestration): M%256==0, N%64==0, Kp%32==0.
// Block = 256 threads (8 wave32); block tile 256x64; per wave 32x64
// = 8 accumulators of v_wmma_f32_16x16x32_f16 per K-step.
// All fragments are preloaded before the WMMA burst -> single s_wait_dscnt.
// ---------------------------------------------------------------------------
__global__ __launch_bounds__(256) void wmma_gemm_kernel(
    const float* __restrict__ A, const _Float16* __restrict__ Wt,
    const float* __restrict__ bias, float* __restrict__ C,
    int M, int N, int Kp, int act)
{
    __shared__ __align__(16) _Float16 As[TILE_M][LDSK];
    __shared__ __align__(16) _Float16 Bs[TILE_N][LDSK];

    const int tid  = threadIdx.x;
    const int lane = tid & 31;
    const int wv   = tid >> 5;
    const int row0 = blockIdx.x * TILE_M;
    const int col0 = blockIdx.y * TILE_N;
    const int half = lane >> 4;
    const int mr   = lane & 15;

    v8f acc[2][4];
    #pragma unroll
    for (int s = 0; s < 2; ++s) { acc[s][0] = {}; acc[s][1] = {}; acc[s][2] = {}; acc[s][3] = {}; }

    // staging assignments: each thread owns one full A row (32 floats / K-step)
    const int bn_ = tid >> 2;           // B col 0..63, 8 halfs each
    const int bkk = (tid & 3) * 8;      // B k offset 0/8/16/24
    const float*    aptr = A  + (size_t)(row0 + tid) * Kp;
    const _Float16* bptr = Wt + (size_t)(col0 + bn_) * Kp + bkk;

    union Frag { v16h v; uint4 q[2]; };

    for (int kc = 0; kc < Kp; kc += TILE_K) {
        // ---- stage A: 8x float4 loads -> 32 f16 -> 4x b128 LDS stores
        {
            const float4* a4 = (const float4*)(aptr + kc);
            union { _Float16 h[32]; uint4 q[4]; } hb;
            #pragma unroll
            for (int u = 0; u < 8; ++u) {
                float4 f = a4[u];
                hb.h[4*u+0] = (_Float16)f.x; hb.h[4*u+1] = (_Float16)f.y;
                hb.h[4*u+2] = (_Float16)f.z; hb.h[4*u+3] = (_Float16)f.w;
            }
            *(uint4*)&As[tid][0]  = hb.q[0];
            *(uint4*)&As[tid][8]  = hb.q[1];
            *(uint4*)&As[tid][16] = hb.q[2];
            *(uint4*)&As[tid][24] = hb.q[3];
            if (kc + TILE_K < Kp)
                __builtin_prefetch(a4 + 8, 0, 0);   // next K-chunk (global_prefetch_b8)
        }
        // ---- stage B: 1x b128 global load -> 1x b128 LDS store
        *(uint4*)&Bs[bn_][bkk] = *(const uint4*)(bptr + kc);
        __syncthreads();

        // ---- preload ALL fragments (12x ds_load_b128), then WMMA burst
        Frag af0, af1, bf[4];
        {
            const _Float16* ar0 = &As[wv * 32 + mr][8 * half];
            const _Float16* ar1 = &As[wv * 32 + 16 + mr][8 * half];
            af0.q[0] = *(const uint4*)ar0;  af0.q[1] = *(const uint4*)(ar0 + 16);
            af1.q[0] = *(const uint4*)ar1;  af1.q[1] = *(const uint4*)(ar1 + 16);
        }
        #pragma unroll
        for (int j = 0; j < 4; ++j) {
            const _Float16* brow = &Bs[j * 16 + mr][8 * half];
            bf[j].q[0] = *(const uint4*)brow;
            bf[j].q[1] = *(const uint4*)(brow + 16);
        }
        #pragma unroll
        for (int j = 0; j < 4; ++j)
            acc[0][j] = __builtin_amdgcn_wmma_f32_16x16x32_f16(
                false, af0.v, false, bf[j].v, (short)0, acc[0][j], false, false);
        #pragma unroll
        for (int j = 0; j < 4; ++j)
            acc[1][j] = __builtin_amdgcn_wmma_f32_16x16x32_f16(
                false, af1.v, false, bf[j].v, (short)0, acc[1][j], false, false);
        __syncthreads();
    }

    // ---- epilogue (unpredicated): m = q + 8*half (+16 for strip 1), n = lane&15
    #pragma unroll
    for (int s = 0; s < 2; ++s) {
        #pragma unroll
        for (int j = 0; j < 4; ++j) {
            const int gn = col0 + j * 16 + mr;
            const float bv = bias[gn];
            float* cp = C + (size_t)(row0 + wv * 32 + s * 16 + 8 * half) * N + gn;
            #pragma unroll
            for (int q = 0; q < 8; ++q) {
                float v = acc[s][j][q] + bv;
                if (act) v = silu_f(v);
                cp[(size_t)q * N] = v;
            }
        }
    }
}

// ---------------------------------------------------------------------------
// weight convert: f32 [K,N] -> f16 transposed+padded [N][Kp]
// ---------------------------------------------------------------------------
__global__ void cvt_wt_kernel(const float* __restrict__ in, _Float16* __restrict__ out,
                              int K, int N, int Kp)
{
    long long i = (long long)blockIdx.x * 256 + threadIdx.x;
    if (i >= (long long)N * Kp) return;
    int n = (int)(i / Kp), k = (int)(i % Kp);
    out[i] = (k < K) ? (_Float16)in[(size_t)k * N + n] : (_Float16)0.0f;
}

__global__ void pe_concat_kernel(const float* __restrict__ x, const float* __restrict__ fx,
                                 const float* __restrict__ Wpe, const float* __restrict__ bpe,
                                 float* __restrict__ hcat, int rows)
{
    size_t i = (size_t)blockIdx.x * 256 + threadIdx.x;
    if (i >= (size_t)rows * 64) return;
    int r = (int)(i >> 6), c = (int)(i & 63);
    float v;
    if (c < 32) {
        const float* xr = x + (size_t)r * 3;
        v = bpe[c] + xr[0] * Wpe[c] + xr[1] * Wpe[32 + c] + xr[2] * Wpe[64 + c];
    } else {
        v = fx[(size_t)r * 32 + (c - 32)];
    }
    hcat[i] = v;
}

// farthest point sampling: one block (256 thr) per batch, dists in registers
__global__ __launch_bounds__(256) void fps_kernel(const float* __restrict__ pos,
                                                  int* __restrict__ out, int n, int m)
{
    const int b = blockIdx.x, tid = threadIdx.x;
    const float* P = pos + (size_t)b * n * 3;
    int* O = out + (size_t)b * m;
    const int chunk = n / 256;
    float dl[16];
    for (int j = 0; j < chunk; ++j) dl[j] = __builtin_inff();
    __shared__ float rv[256]; __shared__ int ri[256];
    __shared__ float lp[3];  __shared__ int slast;
    if (tid == 0) { O[0] = 0; slast = 0; }
    __syncthreads();
    int last = 0;
    for (int it = 1; it < m; ++it) {
        if (tid < 3) lp[tid] = P[(size_t)last * 3 + tid];
        __syncthreads();
        float bv = -1.0f; int bi = 0;
        for (int j = 0; j < chunk; ++j) {
            int i = j * 256 + tid;
            float dx = P[i*3+0]-lp[0], dy = P[i*3+1]-lp[1], dz = P[i*3+2]-lp[2];
            float d = dx*dx + dy*dy + dz*dz;
            float nd = fminf(dl[j], d);
            dl[j] = nd;
            if (nd > bv) { bv = nd; bi = i; }
        }
        rv[tid] = bv; ri[tid] = bi;
        __syncthreads();
        for (int s = 128; s > 0; s >>= 1) {
            if (tid < s) {
                float v2 = rv[tid + s]; int i2 = ri[tid + s];
                if (v2 > rv[tid] || (v2 == rv[tid] && i2 < ri[tid])) { rv[tid] = v2; ri[tid] = i2; }
            }
            __syncthreads();
        }
        if (tid == 0) { O[it] = ri[0]; slast = ri[0]; }
        __syncthreads();
        last = slast;
    }
}

__global__ void gather_pos_kernel(const float* __restrict__ pos, const int* __restrict__ idx,
                                  float* __restrict__ out, int B, int n, int m)
{
    int i = blockIdx.x * 256 + threadIdx.x;
    if (i >= B * m * 3) return;
    int c = i % 3, r = i / 3, b = r / m;
    out[i] = pos[((size_t)b * n + idx[r]) * 3 + c];
}

// K=64 nearest-first selection per query (one block per query), mask = in-radius
__global__ __launch_bounds__(256) void radius_topk_kernel(
    const float* __restrict__ pos, const float* __restrict__ qpos,
    int* __restrict__ nidx, unsigned char* __restrict__ nmask,
    int n, int m, float r2)
{
    __shared__ float d2s[4096];
    __shared__ float rv[256]; __shared__ int ri[256];
    const int q = blockIdx.x, tid = threadIdx.x;
    const int b = q / m;
    const float* qp = qpos + (size_t)q * 3;
    const float qx = qp[0], qy = qp[1], qz = qp[2];
    for (int i = tid; i < n; i += 256) {
        const float* p = pos + ((size_t)b * n + i) * 3;
        float dx = p[0]-qx, dy = p[1]-qy, dz = p[2]-qz;
        d2s[i] = dx*dx + dy*dy + dz*dz;
    }
    __syncthreads();
    for (int kk = 0; kk < 64; ++kk) {
        float best = __builtin_inff(); int bi = n;
        for (int i = tid; i < n; i += 256) {
            float v = d2s[i];
            if (v < best) { best = v; bi = i; }
        }
        rv[tid] = best; ri[tid] = bi;
        __syncthreads();
        for (int s = 128; s > 0; s >>= 1) {
            if (tid < s) {
                float v2 = rv[tid + s]; int i2 = ri[tid + s];
                if (v2 < rv[tid] || (v2 == rv[tid] && i2 < ri[tid])) { rv[tid] = v2; ri[tid] = i2; }
            }
            __syncthreads();
        }
        if (tid == 0) {
            int sel = ri[0];
            nidx[(size_t)q * 64 + kk] = sel;
            nmask[(size_t)q * 64 + kk] = (rv[0] <= r2) ? 1 : 0;
            d2s[sel] = __builtin_inff();
        }
        __syncthreads();
    }
}

// e[row, 0:C] = feat[nidx], e[row, C:C+3] = pos[nidx]-qpos, e[row, C+3:Ke] = 0
__global__ void edge_build_kernel(const float* __restrict__ feat, const float* __restrict__ pos,
                                  const float* __restrict__ qpos, const int* __restrict__ nidx,
                                  float* __restrict__ e, int B, int n, int m, int C, int Ke,
                                  long long tot)
{
    long long i = (long long)blockIdx.x * 256 + threadIdx.x;
    if (i >= tot) return;
    long long row = i / Ke; int c = (int)(i % Ke);
    float v = 0.0f;
    if (c < C + 3) {
        int b = (int)(row / ((long long)m * 64));
        int src = nidx[row];
        if (c < C) v = feat[((size_t)b * n + src) * C + c];
        else {
            int qq = (int)((row / 64) % m);
            int cc = c - C;
            v = pos[((size_t)b * n + src) * 3 + cc] - qpos[((size_t)b * m + qq) * 3 + cc];
        }
    }
    e[i] = v;
}

__global__ void zero_stats_kernel(float* sum, float* sumsq, float* cnt, int N) {
    int i = blockIdx.x * 256 + threadIdx.x;
    if (i < N) { sum[i] = 0.0f; sumsq[i] = 0.0f; }
    if (i == 0) cnt[0] = 0.0f;
}

__global__ __launch_bounds__(256) void bn_reduce_kernel(
    const float* __restrict__ X, const unsigned char* __restrict__ mask,
    float* __restrict__ sum, float* __restrict__ sumsq, float* __restrict__ cnt,
    int M, int N, int stripes)
{
    const int n = blockIdx.x, s = blockIdx.y, tid = threadIdx.x;
    const int rows = (M + stripes - 1) / stripes;
    const int r0 = s * rows;
    int r1 = r0 + rows; if (r1 > M) r1 = M;
    float a = 0.0f, bq = 0.0f, c = 0.0f;
    for (int r = r0 + tid; r < r1; r += 256) {
        float w = mask ? (float)mask[r] : 1.0f;
        float x = X[(size_t)r * N + n];
        a += w * x; bq += w * x * x;
        if (n == 0) c += w;
    }
    __shared__ float sa_[256], sb_[256], sc_[256];
    sa_[tid] = a; sb_[tid] = bq; sc_[tid] = c;
    __syncthreads();
    for (int st = 128; st > 0; st >>= 1) {
        if (tid < st) { sa_[tid] += sa_[tid+st]; sb_[tid] += sb_[tid+st]; sc_[tid] += sc_[tid+st]; }
        __syncthreads();
    }
    if (tid == 0) {
        atomicAdd(&sum[n], sa_[0]);
        atomicAdd(&sumsq[n], sb_[0]);
        if (n == 0) atomicAdd(cnt, sc_[0]);
    }
}

__global__ void bn_finalize_kernel(const float* __restrict__ sum, const float* __restrict__ sumsq,
                                   const float* __restrict__ cnt, const float* __restrict__ g,
                                   const float* __restrict__ be, float* __restrict__ scale,
                                   float* __restrict__ shift, int N)
{
    int n = blockIdx.x * 256 + threadIdx.x;
    if (n >= N) return;
    float wsum = cnt[0];
    float s  = fmaxf(wsum, 1.0f);
    float mu = sum[n] / s;
    float var = (sumsq[n] - 2.0f * mu * sum[n] + mu * mu * wsum) / s;
    float sc = g[n] * rsqrtf(var + 1e-5f);
    scale[n] = sc;
    shift[n] = be[n] - mu * sc;
}

__global__ void bn_act_kernel(float* __restrict__ X, const float* __restrict__ scale,
                              const float* __restrict__ shift, long long tot, int N)
{
    long long i = (long long)blockIdx.x * 256 + threadIdx.x;
    if (i >= tot) return;
    int c = (int)(i % N);
    float y = X[i] * scale[c] + shift[c];
    X[i] = silu_f(y);
}

__global__ void max_agg_kernel(const float* __restrict__ h, const unsigned char* __restrict__ mask,
                               float* __restrict__ out, int rows, int C)
{
    long long i = (long long)blockIdx.x * 256 + threadIdx.x;
    if (i >= (long long)rows * C) return;
    long long row = i / C; int c = (int)(i % C);
    float v = -__builtin_inff();
    for (int k = 0; k < 64; ++k)
        if (mask[row * 64 + k]) v = fmaxf(v, h[((size_t)row * 64 + k) * C + c]);
    out[i] = v;
}

// out[row, 0:C1]=A, [C1:C1+C2]=B, [C1+C2:Kp]=0
__global__ void concat2_kernel(const float* __restrict__ A, const float* __restrict__ Bf,
                               float* __restrict__ out, long long rows, int C1, int C2, int Kp)
{
    long long i = (long long)blockIdx.x * 256 + threadIdx.x;
    long long tot = rows * Kp;
    if (i >= tot) return;
    long long row = i / Kp; int c = (int)(i % Kp);
    float v = 0.0f;
    if (c < C1) v = A[row * C1 + c];
    else if (c < C1 + C2) v = Bf[row * C2 + (c - C1)];
    out[i] = v;
}

__global__ void bcast_concat_kernel(const float* __restrict__ gvec, const float* __restrict__ hf,
                                    float* __restrict__ out, int B, int m, int C1, int C2)
{
    long long i = (long long)blockIdx.x * 256 + threadIdx.x;
    long long tot = (long long)B * m * (C1 + C2);
    if (i >= tot) return;
    int Ct = C1 + C2;
    long long row = i / Ct; int c = (int)(i % Ct);
    int b = (int)(row / m);
    out[i] = (c < C1) ? gvec[(size_t)b * C1 + c] : hf[row * C2 + (c - C1)];
}

__global__ void gpool_kernel(const float* __restrict__ X, float* __restrict__ g, int B, int m, int C)
{
    int i = blockIdx.x * 256 + threadIdx.x;
    if (i >= B * C) return;
    int b = i / C, c = i % C;
    float v = -__builtin_inff();
    for (int j = 0; j < m; ++j) v = fmaxf(v, X[((size_t)b * m + j) * C + c]);
    g[i] = v;
}

__global__ void knn3_kernel(const float* __restrict__ ps, const float* __restrict__ pd,
                            int* __restrict__ idx, float* __restrict__ w, int B, int ns, int nd)
{
    int i = blockIdx.x * 256 + threadIdx.x;
    if (i >= B * nd) return;
    int b = i / nd;
    const float* p = pd + (size_t)i * 3;
    float px = p[0], py = p[1], pz = p[2];
    float d0 = __builtin_inff(), d1 = d0, d2v = d0;
    int i0 = 0, i1 = 0, i2 = 0;
    const float* S = ps + (size_t)b * ns * 3;
    for (int j = 0; j < ns; ++j) {
        float dx = S[j*3+0]-px, dy = S[j*3+1]-py, dz = S[j*3+2]-pz;
        float d = dx*dx + dy*dy + dz*dz;
        if (d < d0)      { d2v = d1; i2 = i1; d1 = d0; i1 = i0; d0 = d; i0 = j; }
        else if (d < d1) { d2v = d1; i2 = i1; d1 = d;  i1 = j; }
        else if (d < d2v){ d2v = d;  i2 = j; }
    }
    float w0 = 1.0f / fmaxf(d0, 1e-16f);
    float w1 = 1.0f / fmaxf(d1, 1e-16f);
    float w2 = 1.0f / fmaxf(d2v, 1e-16f);
    float inv = 1.0f / (w0 + w1 + w2);
    idx[(size_t)i*3+0] = i0; idx[(size_t)i*3+1] = i1; idx[(size_t)i*3+2] = i2;
    w[(size_t)i*3+0] = w0*inv; w[(size_t)i*3+1] = w1*inv; w[(size_t)i*3+2] = w2*inv;
}

__global__ void interp3_kernel(const float* __restrict__ xs, const int* __restrict__ kidx,
                               const float* __restrict__ kw, float* __restrict__ out,
                               int B, int ns, int nd, int C)
{
    long long i = (long long)blockIdx.x * 256 + threadIdx.x;
    long long tot = (long long)B * nd * C;
    if (i >= tot) return;
    int c = (int)(i % C);
    long long dst = i / C;
    int b = (int)(dst / nd);
    float acc = 0.0f;
    #pragma unroll
    for (int k = 0; k < 3; ++k)
        acc += kw[dst*3+k] * xs[((size_t)b * ns + kidx[dst*3+k]) * C + c];
    out[i] = acc;
}

__global__ void proj2_kernel(const float* __restrict__ H, const float* __restrict__ W,
                             const float* __restrict__ bb, float* __restrict__ out, int rows)
{
    int i = blockIdx.x * 256 + threadIdx.x;
    if (i >= rows * 3) return;
    int r = i / 3, c = i % 3;
    const float* h = H + (size_t)r * 128;
    float acc = bb[c];
    for (int k = 0; k < 128; ++k) acc += h[k] * W[k * 3 + c];
    out[i] = acc;
}

// ---------------------------------------------------------------------------
// host orchestration
// ---------------------------------------------------------------------------
extern "C" void kernel_launch(void* const* d_in, const int* in_sizes, int n_in,
                              void* d_out, int out_size, void* d_ws, size_t ws_size,
                              hipStream_t stream)
{
    (void)in_sizes; (void)n_in; (void)out_size; (void)ws_size;
    const int B = 8, N = 4096, M1 = 1024, M2 = 256, KN = 64;
    const float R1sq = 0.04f, R2sq = 0.16f;

    const float* x   = (const float*)d_in[0];
    const float* fx  = (const float*)d_in[1];
    const float* peW = (const float*)d_in[2];
    const float* peB = (const float*)d_in[3];
    const float* lB1 = (const float*)d_in[5];
    const float* lB2 = (const float*)d_in[7];
    const int SA1 = 8, SA2 = 20, SA3 = 32, FP3 = 44, FP2 = 52, FP1 = 60;
    auto LW = [&](int base, int li, int comp) { return (const float*)d_in[base + li*4 + comp]; };
    const float* pB1 = (const float*)d_in[69];
    const float* pW2 = (const float*)d_in[70];
    const float* pB2 = (const float*)d_in[71];

    char* ws = (char*)d_ws;
    size_t off = 0;
    auto alloc = [&](size_t bytes) -> void* {
        void* p = ws + off;
        off += (bytes + 255) & ~(size_t)255;
        return p;
    };
    auto g1d = [](long long tot) { return dim3((unsigned)((tot + 255) / 256)); };

    // f16 weight copies (transposed [N][Kp], K zero-padded) for WMMA B operands
    struct WRec { const float* W; int K, N, Kp; _Float16* Wh; };
    WRec wr[18] = {
        {(const float*)d_in[4], 64, 128, 0, nullptr}, {(const float*)d_in[6], 128, 128, 0, nullptr},
        {LW(SA1,0,0),131,64,0,nullptr},{LW(SA1,1,0),64,64,0,nullptr},{LW(SA1,2,0),64,128,0,nullptr},
        {LW(SA2,0,0),131,64,0,nullptr},{LW(SA2,1,0),64,64,0,nullptr},{LW(SA2,2,0),64,256,0,nullptr},
        {LW(SA3,0,0),259,256,0,nullptr},{LW(SA3,1,0),256,512,0,nullptr},{LW(SA3,2,0),512,1024,0,nullptr},
        {LW(FP3,0,0),1280,256,0,nullptr},{LW(FP3,1,0),256,256,0,nullptr},
        {LW(FP2,0,0),384,128,0,nullptr},{LW(FP2,1,0),128,128,0,nullptr},
        {LW(FP1,0,0),256,128,0,nullptr},{LW(FP1,1,0),128,128,0,nullptr},
        {(const float*)d_in[68],128,128,0,nullptr}
    };
    for (int i = 0; i < 18; ++i) {
        wr[i].Kp = (wr[i].K + 31) & ~31;
        long long tot = (long long)wr[i].N * wr[i].Kp;
        wr[i].Wh = (_Float16*)alloc((size_t)tot * sizeof(_Float16));
        cvt_wt_kernel<<<g1d(tot), 256, 0, stream>>>(wr[i].W, wr[i].Wh, wr[i].K, wr[i].N, wr[i].Kp);
    }

    // persistent buffers
    float* hcat  = (float*)alloc((size_t)B*N*64*4);
    float* lifth = (float*)alloc((size_t)B*N*128*4);
    float* h     = (float*)alloc((size_t)B*N*128*4);
    int*   idx1  = (int*)alloc((size_t)B*M1*4);
    float* pos1  = (float*)alloc((size_t)B*M1*3*4);
    int*   ni1   = (int*)alloc((size_t)B*M1*KN*4);
    unsigned char* nm1 = (unsigned char*)alloc((size_t)B*M1*KN);
    float* h1    = (float*)alloc((size_t)B*M1*128*4);
    int*   idx2  = (int*)alloc((size_t)B*M2*4);
    float* pos2  = (float*)alloc((size_t)B*M2*3*4);
    int*   ni2   = (int*)alloc((size_t)B*M2*KN*4);
    unsigned char* nm2 = (unsigned char*)alloc((size_t)B*M2*KN);
    float* h2    = (float*)alloc((size_t)B*M2*256*4);
    float* gv    = (float*)alloc((size_t)B*1024*4);
    float* f3    = (float*)alloc((size_t)B*M2*256*4);
    float* f2    = (float*)alloc((size_t)B*M1*128*4);
    float* f1    = (float*)alloc((size_t)B*N*128*4);
    float* ssum  = (float*)alloc(1024*4);
    float* ssq   = (float*)alloc(1024*4);
    float* scnt  = (float*)alloc(256);
    float* sscale= (float*)alloc(1024*4);
    float* sshift= (float*)alloc(1024*4);
    int*   kidx  = (int*)alloc((size_t)B*N*3*4);
    float* kwt   = (float*)alloc((size_t)B*N*3*4);
    const size_t stage_base = off;

    auto gemm = [&](const float* A, const WRec& w, const float* bias, float* C,
                    int M, int act) {
        dim3 grid(M / TILE_M, w.N / TILE_N);
        wmma_gemm_kernel<<<grid, 256, 0, stream>>>(A, w.Wh, bias, C, M, w.N, w.Kp, act);
    };
    auto bn_silu = [&](float* X, const float* gg, const float* be, int M, int Nc,
                       const unsigned char* mask) {
        zero_stats_kernel<<<g1d(Nc), 256, 0, stream>>>(ssum, ssq, scnt, Nc);
        int stripes = M / 1024; if (stripes < 1) stripes = 1; if (stripes > 64) stripes = 64;
        bn_reduce_kernel<<<dim3(Nc, stripes), 256, 0, stream>>>(X, mask, ssum, ssq, scnt, M, Nc, stripes);
        bn_finalize_kernel<<<g1d(Nc), 256, 0, stream>>>(ssum, ssq, scnt, gg, be, sscale, sshift, Nc);
        long long tot = (long long)M * Nc;
        bn_act_kernel<<<g1d(tot), 256, 0, stream>>>(X, sscale, sshift, tot, Nc);
    };

    // ---- pe + lift ----
    pe_concat_kernel<<<g1d((long long)B*N*64), 256, 0, stream>>>(x, fx, peW, peB, hcat, B*N);
    gemm(hcat,  wr[0], lB1, lifth, B*N, 1);   // fused SiLU
    gemm(lifth, wr[1], lB2, h,     B*N, 0);

    // ---- SA1 ----
    fps_kernel<<<B, 256, 0, stream>>>(x, idx1, N, M1);
    gather_pos_kernel<<<g1d(B*M1*3), 256, 0, stream>>>(x, idx1, pos1, B, N, M1);
    radius_topk_kernel<<<B*M1, 256, 0, stream>>>(x, pos1, ni1, nm1, N, M1, R1sq);
    {
        off = stage_base;
        const int E = B*M1*KN, Ke = 160;     // 131 -> padded 160
        float* e1  = (float*)alloc((size_t)E*Ke*4);
        float* t1a = (float*)alloc((size_t)E*64*4);
        float* t1b = (float*)alloc((size_t)E*64*4);
        float* t1c = (float*)alloc((size_t)E*128*4);
        long long etot = (long long)E*Ke;
        edge_build_kernel<<<g1d(etot), 256, 0, stream>>>(h, x, pos1, ni1, e1, B, N, M1, 128, Ke, etot);
        gemm(e1,  wr[2], LW(SA1,0,1), t1a, E, 0);  bn_silu(t1a, LW(SA1,0,2), LW(SA1,0,3), E, 64, nm1);
        gemm(t1a, wr[3], LW(SA1,1,1), t1b, E, 0);  bn_silu(t1b, LW(SA1,1,2), LW(SA1,1,3), E, 64, nm1);
        gemm(t1b, wr[4], LW(SA1,2,1), t1c, E, 0);  bn_silu(t1c, LW(SA1,2,2), LW(SA1,2,3), E, 128, nm1);
        max_agg_kernel<<<g1d((long long)B*M1*128), 256, 0, stream>>>(t1c, nm1, h1, B*M1, 128);
    }

    // ---- SA2 ----
    fps_kernel<<<B, 256, 0, stream>>>(pos1, idx2, M1, M2);
    gather_pos_kernel<<<g1d(B*M2*3), 256, 0, stream>>>(pos1, idx2, pos2, B, M1, M2);
    radius_topk_kernel<<<B*M2, 256, 0, stream>>>(pos1, pos2, ni2, nm2, M1, M2, R2sq);
    {
        off = stage_base;
        const int E = B*M2*KN, Ke = 160;
        float* e2  = (float*)alloc((size_t)E*Ke*4);
        float* t2a = (float*)alloc((size_t)E*64*4);
        float* t2b = (float*)alloc((size_t)E*64*4);
        float* t2c = (float*)alloc((size_t)E*256*4);
        long long etot = (long long)E*Ke;
        edge_build_kernel<<<g1d(etot), 256, 0, stream>>>(h1, pos1, pos2, ni2, e2, B, M1, M2, 128, Ke, etot);
        gemm(e2,  wr[5], LW(SA2,0,1), t2a, E, 0);  bn_silu(t2a, LW(SA2,0,2), LW(SA2,0,3), E, 64, nm2);
        gemm(t2a, wr[6], LW(SA2,1,1), t2b, E, 0);  bn_silu(t2b, LW(SA2,1,2), LW(SA2,1,3), E, 64, nm2);
        gemm(t2b, wr[7], LW(SA2,2,1), t2c, E, 0);  bn_silu(t2c, LW(SA2,2,2), LW(SA2,2,3), E, 256, nm2);
        max_agg_kernel<<<g1d((long long)B*M2*256), 256, 0, stream>>>(t2c, nm2, h2, B*M2, 256);
    }

    // ---- SA3 (global) ----
    {
        off = stage_base;
        const int R = B*M2, Ke = 288;        // 259 -> padded 288
        float* e3  = (float*)alloc((size_t)R*Ke*4);
        float* t3a = (float*)alloc((size_t)R*256*4);
        float* t3b = (float*)alloc((size_t)R*512*4);
        float* t3c = (float*)alloc((size_t)R*1024*4);
        concat2_kernel<<<g1d((long long)R*Ke), 256, 0, stream>>>(h2, pos2, e3, R, 256, 3, Ke);
        gemm(e3,  wr[8],  LW(SA3,0,1), t3a, R, 0);  bn_silu(t3a, LW(SA3,0,2), LW(SA3,0,3), R, 256, nullptr);
        gemm(t3a, wr[9],  LW(SA3,1,1), t3b, R, 0);  bn_silu(t3b, LW(SA3,1,2), LW(SA3,1,3), R, 512, nullptr);
        gemm(t3b, wr[10], LW(SA3,2,1), t3c, R, 0);  bn_silu(t3c, LW(SA3,2,2), LW(SA3,2,3), R, 1024, nullptr);
        gpool_kernel<<<g1d(B*1024), 256, 0, stream>>>(t3c, gv, B, M2, 1024);

        // ---- FP3 (k=1: broadcast of global vector) ----
        off = stage_base;
        float* f3in = (float*)alloc((size_t)R*1280*4);
        float* f3a  = (float*)alloc((size_t)R*256*4);
        bcast_concat_kernel<<<g1d((long long)R*1280), 256, 0, stream>>>(gv, h2, f3in, B, M2, 1024, 256);
        gemm(f3in, wr[11], LW(FP3,0,1), f3a, R, 0); bn_silu(f3a, LW(FP3,0,2), LW(FP3,0,3), R, 256, nullptr);
        gemm(f3a,  wr[12], LW(FP3,1,1), f3,  R, 0); bn_silu(f3,  LW(FP3,1,2), LW(FP3,1,3), R, 256, nullptr);
    }

    // ---- FP2 (k=3 interp pos2 -> pos1) ----
    knn3_kernel<<<g1d(B*M1), 256, 0, stream>>>(pos2, pos1, kidx, kwt, B, M2, M1);
    {
        off = stage_base;
        const int R = B*M1;
        float* xi2  = (float*)alloc((size_t)R*256*4);
        float* f2in = (float*)alloc((size_t)R*384*4);
        float* f2a  = (float*)alloc((size_t)R*128*4);
        interp3_kernel<<<g1d((long long)R*256), 256, 0, stream>>>(f3, kidx, kwt, xi2, B, M2, M1, 256);
        concat2_kernel<<<g1d((long long)R*384), 256, 0, stream>>>(xi2, h1, f2in, R, 256, 128, 384);
        gemm(f2in, wr[13], LW(FP2,0,1), f2a, R, 0); bn_silu(f2a, LW(FP2,0,2), LW(FP2,0,3), R, 128, nullptr);
        gemm(f2a,  wr[14], LW(FP2,1,1), f2,  R, 0); bn_silu(f2,  LW(FP2,1,2), LW(FP2,1,3), R, 128, nullptr);
    }

    // ---- FP1 (k=3 interp pos1 -> pos) ----
    knn3_kernel<<<g1d(B*N), 256, 0, stream>>>(pos1, x, kidx, kwt, B, M1, N);
    {
        off = stage_base;
        const int R = B*N;
        float* xi1  = (float*)alloc((size_t)R*128*4);
        float* f1in = (float*)alloc((size_t)R*256*4);
        float* f1a  = (float*)alloc((size_t)R*128*4);
        interp3_kernel<<<g1d((long long)R*128), 256, 0, stream>>>(f2, kidx, kwt, xi1, B, M1, N, 128);
        concat2_kernel<<<g1d((long long)R*256), 256, 0, stream>>>(xi1, h, f1in, R, 128, 128, 256);
        gemm(f1in, wr[15], LW(FP1,0,1), f1a, R, 0); bn_silu(f1a, LW(FP1,0,2), LW(FP1,0,3), R, 128, nullptr);
        gemm(f1a,  wr[16], LW(FP1,1,1), f1,  R, 0); bn_silu(f1,  LW(FP1,1,2), LW(FP1,1,3), R, 128, nullptr);
    }

    // ---- projection head ----
    {
        off = stage_base;
        const int R = B*N;
        float* ph = (float*)alloc((size_t)R*128*4);
        gemm(f1, wr[17], pB1, ph, R, 1);  // fused SiLU
        proj2_kernel<<<g1d((long long)R*3), 256, 0, stream>>>(ph, pW2, pB2, (float*)d_out, R);
    }
}
